// MixtureOfExperts_45698452029483
// MI455X (gfx1250) — matrix operand order repaired and checked by reading, not compiled, not run on hardware
//
#include <hip/hip_runtime.h>
#include <hip/hip_bf16.h>

// ---------------------------------------------------------------------------
// MoE (router top-2 + expert FFN + residual + LN) for gfx1250 (MI455X)
// Routed implementation (only the 2 selected experts per token computed).
// bf16 WMMA 16x16x32 with fp32 accumulation; weights pre-converted to bf16
// in a WMMA-B-fragment tiled layout (each fragment = 2 contiguous b128 loads).
// ---------------------------------------------------------------------------

typedef __bf16 bf16_t;
typedef __attribute__((ext_vector_type(16))) __bf16 v16bf;
typedef __attribute__((ext_vector_type(8)))  float  v8f;

#define D_MODEL   1024
#define D_FF      4096
#define N_EXPERTS 8
#define N_TOKENS  4096   // 2 * 2048
#define TILE_M    32     // tokens per workgroup
#define FC        512    // F-chunk width
#define NCHUNK    (D_FF / FC)   // 8
#define XS        1032   // LDS x row stride (bf16 elems), padded (bank-friendly)
#define HS        520    // LDS h row stride (bf16 elems), padded
#define FRAGS_PER_EXPERT 8192   // (K/32)*(N/16) for both w1 and w2
#define LN_EPS    1e-6f

union FragU { uint4 u[2]; v16bf v; bf16_t h[16]; };

__device__ __forceinline__ v8f wmma_bf16(v16bf a, v16bf b, v8f c) {
  // D = A(16x32 bf16) * B(32x16 bf16) + C(16x16 f32)
  return __builtin_amdgcn_wmma_f32_16x16x32_bf16(false, a, false, b,
                                                 (short)0, c, false, false);
}

// ---------------------------------------------------------------------------
// Kernel 0: zero per-expert counters
// ---------------------------------------------------------------------------
__global__ void moe_zero_cnt(int* __restrict__ cnt) {
  if (threadIdx.x < N_EXPERTS) cnt[threadIdx.x] = 0;
}

// ---------------------------------------------------------------------------
// Kernel 1: convert fp32 weights [E][K][N] -> bf16 WMMA-B fragment tiles
// layout: [e][kb=K/32][nb=N/16][lane 0..31][16 bf16]   (512 bf16 per frag)
// lane L holds column n = nb*16 + (L%16), K run = kb*32 + (L/16)*16 .. +15
// ---------------------------------------------------------------------------
__global__ void __launch_bounds__(256)
moe_convert_kernel(const float* __restrict__ src, bf16_t* __restrict__ dst,
                   int K, int N, int nbTot) {
  int wid  = threadIdx.x >> 5;
  int lane = threadIdx.x & 31;
  int fb   = blockIdx.x * 8 + wid;              // global fragment id
  int e    = fb / FRAGS_PER_EXPERT;
  int rem  = fb - e * FRAGS_PER_EXPERT;
  int kb   = rem / nbTot;
  int nb   = rem - kb * nbTot;
  int n     = nb * 16 + (lane & 15);
  int kbase = kb * 32 + (lane >> 4) * 16;
  const float* s = src + ((size_t)e * K + kbase) * N + n;
  FragU f;
#pragma unroll
  for (int j = 0; j < 16; ++j) f.h[j] = (bf16_t)s[(size_t)j * N];
  bf16_t* d = dst + ((size_t)fb * 32 + lane) * 16;
  *(uint4*)d       = f.u[0];
  *(uint4*)(d + 8) = f.u[1];
}

// ---------------------------------------------------------------------------
// Kernel 2: router. One wave per token: logits = x . router_w, top-2,
// softmax over the 2 selected, append (token, gate) to expert list.
// ---------------------------------------------------------------------------
__global__ void __launch_bounds__(256)
moe_router_kernel(const float* __restrict__ x, const float* __restrict__ rw,
                  int* __restrict__ cnt, int* __restrict__ eTok,
                  float* __restrict__ eGate, int* __restrict__ tokslot) {
  int wid  = threadIdx.x >> 5;
  int lane = threadIdx.x & 31;
  int t = blockIdx.x * 8 + wid;
  float acc[N_EXPERTS];
#pragma unroll
  for (int e = 0; e < N_EXPERTS; ++e) acc[e] = 0.0f;
  const float4* rw4 = (const float4*)rw;   // [1024][8] -> 2 float4 per row
  for (int k = lane; k < D_MODEL; k += 32) {
    float xv = x[(size_t)t * D_MODEL + k];
    float4 a = rw4[k * 2 + 0];
    float4 b = rw4[k * 2 + 1];
    acc[0] += xv * a.x; acc[1] += xv * a.y; acc[2] += xv * a.z; acc[3] += xv * a.w;
    acc[4] += xv * b.x; acc[5] += xv * b.y; acc[6] += xv * b.z; acc[7] += xv * b.w;
  }
#pragma unroll
  for (int e = 0; e < N_EXPERTS; ++e) {
#pragma unroll
    for (int m = 16; m >= 1; m >>= 1) acc[e] += __shfl_xor(acc[e], m, 32);
  }
  if (lane == 0) {
    int i0 = 0; float v0 = acc[0];
#pragma unroll
    for (int e = 1; e < N_EXPERTS; ++e)
      if (acc[e] > v0) { v0 = acc[e]; i0 = e; }
    int i1 = (i0 == 0) ? 1 : 0; float v1 = acc[i1];
#pragma unroll
    for (int e = 0; e < N_EXPERTS; ++e)
      if (e != i0 && acc[e] > v1) { v1 = acc[e]; i1 = e; }
    float g0 = 1.0f / (1.0f + expf(v1 - v0));   // softmax over {v0, v1}
    float g1 = 1.0f - g0;
    int p0 = atomicAdd(&cnt[i0], 1);
    eTok[i0 * N_TOKENS + p0]  = t;
    eGate[i0 * N_TOKENS + p0] = g0;
    tokslot[2 * t + 0] = i0 * N_TOKENS + p0;
    int p1 = atomicAdd(&cnt[i1], 1);
    eTok[i1 * N_TOKENS + p1]  = t;
    eGate[i1 * N_TOKENS + p1] = g1;
    tokslot[2 * t + 1] = i1 * N_TOKENS + p1;
  }
}

// ---------------------------------------------------------------------------
// Kernel 3: expert FFN. Grid (tile, expert); block = 512 threads (16 waves).
// 32 tokens x full FFN: h = gelu(x@w1+b1); out = h@w2; res+b2; LN; gate.
// GEMM1: wave w owns w1 column tiles {2w, 2w+1} of the 512-wide chunk
//        -> 2x2 C tiles per wave, 2 mem ops per wmma.
// GEMM2: wave w owns output columns [64w, 64w+64): 8 fp32 C tiles (2r x 4c).
// ---------------------------------------------------------------------------
__global__ void __launch_bounds__(512)
moe_ffn_kernel(const float* __restrict__ x,
               const float* __restrict__ b1,
               const float* __restrict__ b2,
               const float* __restrict__ gamma,
               const float* __restrict__ beta,
               const bf16_t* __restrict__ w1t,
               const bf16_t* __restrict__ w2t,
               const int* __restrict__ cnt,
               const int* __restrict__ eTok,
               const float* __restrict__ eGate,
               float* __restrict__ slotbuf) {
  int e    = blockIdx.y;
  int tile = blockIdx.x;
  int count = cnt[e];
  if (tile * TILE_M >= count) return;

  extern __shared__ char smem[];
  bf16_t* xbuf   = (bf16_t*)smem;                                   // 32*XS bf16
  bf16_t* hbuf   = (bf16_t*)(smem + 32 * XS * 2);                   // 32*HS bf16
  float*  rowsum = (float*)(smem + 32 * XS * 2 + 32 * HS * 2);      // 32*16
  float*  rowsq  = rowsum + 32 * 16;                                // 32*16
  float*  muL    = rowsq + 32 * 16;                                 // 32
  float*  rsL    = muL + 32;                                        // 32
  int*    tokL   = (int*)(rsL + 32);                                // 32
  float*  gateL  = (float*)(tokL + 32);                             // 32

  int tid   = threadIdx.x;
  int wid   = tid >> 5;
  int lane  = tid & 31;
  int lhalf = lane >> 4;
  int lcol  = lane & 15;

  if (tid < TILE_M) {
    int g = tile * TILE_M + tid;
    bool valid = g < count;
    tokL[tid]  = valid ? eTok[e * N_TOKENS + g] : 0;
    gateL[tid] = valid ? eGate[e * N_TOKENS + g] : 0.0f;
  }
  __syncthreads();

  // gather x rows (fp32 -> bf16) into LDS, 2 rows per wave
#pragma unroll
  for (int rr = 0; rr < 2; ++rr) {
    int r = wid * 2 + rr;
    const float4* xr = (const float4*)(x + (size_t)tokL[r] * D_MODEL);
#pragma unroll
    for (int b = 0; b < 8; ++b) {
      float4 v = xr[lane + b * 32];
      union { bf16_t h[4]; uint2 u; } p;
      p.h[0] = (bf16_t)v.x; p.h[1] = (bf16_t)v.y;
      p.h[2] = (bf16_t)v.z; p.h[3] = (bf16_t)v.w;
      *(uint2*)&xbuf[r * XS + lane * 4 + b * 128] = p.u;
    }
  }
  __syncthreads();

  const bf16_t* w1e = w1t + (size_t)e * FRAGS_PER_EXPERT * 512;
  const bf16_t* w2e = w2t + (size_t)e * FRAGS_PER_EXPERT * 512;

  v8f zero8 = {0.f, 0.f, 0.f, 0.f, 0.f, 0.f, 0.f, 0.f};
  v8f oC[8];
#pragma unroll
  for (int t = 0; t < 8; ++t) oC[t] = zero8;

  for (int c = 0; c < NCHUNK; ++c) {
    // ---- GEMM1: h_chunk(32 x 512) = x(32 x 1024) @ w1[:, chunk] ----
    v8f hC[4];                               // [rt][jj] -> rt*2+jj
#pragma unroll
    for (int t = 0; t < 4; ++t) hC[t] = zero8;
    int nbG0 = c * 32 + wid * 2;             // this wave's first w1 column tile
#pragma unroll 2
    for (int kb = 0; kb < 32; ++kb) {
      FragU aF[2];
#pragma unroll
      for (int rt = 0; rt < 2; ++rt) {
        const bf16_t* ap = &xbuf[(rt * 16 + lcol) * XS + kb * 32 + lhalf * 16];
        aF[rt].u[0] = *(const uint4*)ap;
        aF[rt].u[1] = *(const uint4*)(ap + 8);
      }
#pragma unroll
      for (int jj = 0; jj < 2; ++jj) {
        FragU bF;
        const bf16_t* bp =
            w1e + ((size_t)(kb * 256 + nbG0 + jj)) * 512 + lane * 16;
        bF.u[0] = *(const uint4*)bp;
        bF.u[1] = *(const uint4*)(bp + 8);
#pragma unroll
        for (int rt = 0; rt < 2; ++rt)
          hC[rt * 2 + jj] = wmma_bf16(aF[rt].v, bF.v, hC[rt * 2 + jj]);
      }
    }
    __syncthreads();   // previous chunk's GEMM2 done reading hbuf
#pragma unroll
    for (int jj = 0; jj < 2; ++jj) {
      int fcol = (wid * 2 + jj) * 16 + lcol;           // column within chunk
      float b1v = b1[e * D_FF + c * FC + fcol];
#pragma unroll
      for (int rt = 0; rt < 2; ++rt) {
#pragma unroll
        for (int i = 0; i < 8; ++i) {
          float v = hC[rt * 2 + jj][i] + b1v;
          v = 0.5f * v * (1.0f + erff(v * 0.70710678118f));   // exact gelu
          hbuf[(rt * 16 + i + 8 * lhalf) * HS + fcol] = (bf16_t)v;
        }
      }
    }
    __syncthreads();   // hbuf ready

    // ---- GEMM2: out(32 x 1024) += h_chunk(32 x 512) @ w2[chunk, :] ----
#pragma unroll 1
    for (int kk = 0; kk < 16; ++kk) {
      FragU aF[2];
#pragma unroll
      for (int rt = 0; rt < 2; ++rt) {
        const bf16_t* ap = &hbuf[(rt * 16 + lcol) * HS + kk * 32 + lhalf * 16];
        aF[rt].u[0] = *(const uint4*)ap;
        aF[rt].u[1] = *(const uint4*)(ap + 8);
      }
#pragma unroll
      for (int j = 0; j < 4; ++j) {
        FragU bF;
        const bf16_t* bp =
            w2e + ((size_t)((c * 16 + kk) * 64 + wid * 4 + j)) * 512 + lane * 16;
        bF.u[0] = *(const uint4*)bp;
        bF.u[1] = *(const uint4*)(bp + 8);
#pragma unroll
        for (int rt = 0; rt < 2; ++rt)
          oC[rt * 4 + j] = wmma_bf16(aF[rt].v, bF.v, oC[rt * 4 + j]);
      }
    }
  }

  // ---- residual + b2, LayerNorm, gate, store to slot buffer ----
  float b2v[4], gv[4], bv[4];
#pragma unroll
  for (int j = 0; j < 4; ++j) {
    int col = wid * 64 + j * 16 + lcol;
    b2v[j] = b2[e * D_MODEL + col];
    gv[j]  = gamma[e * D_MODEL + col];
    bv[j]  = beta[e * D_MODEL + col];
  }
#pragma unroll
  for (int rt = 0; rt < 2; ++rt) {
#pragma unroll
    for (int i = 0; i < 8; ++i) {
      int row = rt * 16 + i + 8 * lhalf;
      const float* xr = x + (size_t)tokL[row] * D_MODEL;
#pragma unroll
      for (int j = 0; j < 4; ++j) {
        int col = wid * 64 + j * 16 + lcol;
        oC[rt * 4 + j][i] += xr[col] + b2v[j];
      }
    }
  }
  // per-row partial sums over this wave's 64 columns
#pragma unroll
  for (int rt = 0; rt < 2; ++rt) {
#pragma unroll
    for (int i = 0; i < 8; ++i) {
      float s = 0.f, q = 0.f;
#pragma unroll
      for (int j = 0; j < 4; ++j) { float v = oC[rt * 4 + j][i]; s += v; q += v * v; }
#pragma unroll
      for (int m = 1; m < 16; m <<= 1) {   // reduce within 16-lane half (same row)
        s += __shfl_xor(s, m, 32);
        q += __shfl_xor(q, m, 32);
      }
      if (lcol == 0) {
        int row = rt * 16 + i + 8 * lhalf;
        rowsum[row * 16 + wid] = s;
        rowsq[row * 16 + wid]  = q;
      }
    }
  }
  __syncthreads();
  if (tid < TILE_M) {
    float s = 0.f, q = 0.f;
#pragma unroll
    for (int w = 0; w < 16; ++w) { s += rowsum[tid * 16 + w]; q += rowsq[tid * 16 + w]; }
    float mu  = s * (1.0f / (float)D_MODEL);
    float var = q * (1.0f / (float)D_MODEL) - mu * mu;
    muL[tid] = mu;
    rsL[tid] = rsqrtf(var + LN_EPS);
  }
  __syncthreads();
  size_t slotBase = (size_t)e * N_TOKENS + (size_t)tile * TILE_M;
#pragma unroll
  for (int rt = 0; rt < 2; ++rt) {
#pragma unroll
    for (int i = 0; i < 8; ++i) {
      int row = rt * 16 + i + 8 * lhalf;
      float mu = muL[row], rs = rsL[row], gt = gateL[row];
      size_t base = (slotBase + row) * D_MODEL;
#pragma unroll
      for (int j = 0; j < 4; ++j) {
        int col = wid * 64 + j * 16 + lcol;
        float v = (oC[rt * 4 + j][i] - mu) * rs * gv[j] + bv[j];
        slotbuf[base + col] = gt * v;
      }
    }
  }
}

// ---------------------------------------------------------------------------
// Kernel 4: combine per-token from its 2 assignment slots
// ---------------------------------------------------------------------------
__global__ void __launch_bounds__(256)
moe_combine_kernel(const float* __restrict__ slotbuf,
                   const int* __restrict__ tokslot, float* __restrict__ y) {
  int idx = blockIdx.x * 256 + threadIdx.x;
  if (idx >= N_TOKENS * D_MODEL) return;
  int n = idx >> 10;
  int d = idx & (D_MODEL - 1);
  int s0 = tokslot[2 * n + 0];
  int s1 = tokslot[2 * n + 1];
  y[idx] = slotbuf[(size_t)s0 * D_MODEL + d] + slotbuf[(size_t)s1 * D_MODEL + d];
}

// ---------------------------------------------------------------------------
// Host launch
// ---------------------------------------------------------------------------
extern "C" void kernel_launch(void* const* d_in, const int* in_sizes, int n_in,
                              void* d_out, int out_size, void* d_ws, size_t ws_size,
                              hipStream_t stream) {
  const float* x        = (const float*)d_in[0];
  const float* router_w = (const float*)d_in[1];
  const float* w1       = (const float*)d_in[2];
  const float* b1       = (const float*)d_in[3];
  const float* w2       = (const float*)d_in[4];
  const float* b2       = (const float*)d_in[5];
  const float* gamma    = (const float*)d_in[6];
  const float* beta     = (const float*)d_in[7];
  float* y = (float*)d_out;

  // workspace carve-out
  char* ws = (char*)d_ws;
  size_t off = 0;
  auto carve = [&](size_t bytes) -> char* {
    char* p = ws + off;
    off += (bytes + 255) & ~(size_t)255;
    return p;
  };
  const size_t WT_BYTES = (size_t)N_EXPERTS * 1024 * 4096 * sizeof(bf16_t); // 64 MiB
  bf16_t* w1t    = (bf16_t*)carve(WT_BYTES);
  bf16_t* w2t    = (bf16_t*)carve(WT_BYTES);
  int*    cnt    = (int*)carve(N_EXPERTS * sizeof(int));
  int*    eTok   = (int*)carve((size_t)N_EXPERTS * N_TOKENS * sizeof(int));
  float*  eGate  = (float*)carve((size_t)N_EXPERTS * N_TOKENS * sizeof(float));
  int*    tokslot = (int*)carve((size_t)N_TOKENS * 2 * sizeof(int));
  float*  slotbuf = (float*)carve((size_t)N_EXPERTS * N_TOKENS * D_MODEL * sizeof(float));
  (void)ws_size; (void)in_sizes; (void)n_in; (void)out_size;

  // 0) zero expert counters
  moe_zero_cnt<<<1, 32, 0, stream>>>(cnt);

  // 1) weights -> bf16 fragment tiles (65536 fragment blocks each, 8/block)
  moe_convert_kernel<<<FRAGS_PER_EXPERT, 256, 0, stream>>>(w1, w1t, 1024, 4096, 256);
  moe_convert_kernel<<<FRAGS_PER_EXPERT, 256, 0, stream>>>(w2, w2t, 4096, 1024, 64);

  // 2) router (one wave per token; 8 waves / block)
  moe_router_kernel<<<N_TOKENS / 8, 256, 0, stream>>>(x, router_w, cnt, eTok,
                                                      eGate, tokslot);

  // 3) expert FFN (early-exit on tiles beyond routed count)
  int smemBytes = 32 * XS * 2 + 32 * HS * 2      // xbuf + hbuf (bf16)
                + 32 * 16 * 4 * 2                // rowsum + rowsq
                + 32 * 4 * 2                     // mu + rstd
                + 32 * 4 + 32 * 4;               // tok + gate
  dim3 ffnGrid(N_TOKENS / TILE_M, N_EXPERTS);
  moe_ffn_kernel<<<ffnGrid, 512, smemBytes, stream>>>(
      x, b1, b2, gamma, beta, w1t, w2t, cnt, eTok, eGate, slotbuf);

  // 4) combine the two expert contributions per token
  moe_combine_kernel<<<(N_TOKENS * D_MODEL) / 256, 256, 0, stream>>>(slotbuf,
                                                                    tokslot, y);
}
